// GCN_59846074302535
// MI455X (gfx1250) — matrix operand order repaired and checked by reading, not compiled, not run on hardware
//
#include <hip/hip_runtime.h>
#include <hip/hip_bf16.h>

// GCN 2-layer inference for MI455X (gfx1250, wave32, WMMA).
// Dense transforms: v_wmma_f32_16x16x32_bf16, f32 accumulate.
// B matrix staged in LDS pre-swizzled to the WMMA B-fragment layout and
// hoisted to registers; A fragments are 2x ds_load_b128; C tile staged
// through LDS for coalesced b128 global stores. All 32-bit indexing.
// Edge aggregation: float4 gathers + global f32 atomics.

#define IN_F   128
#define HID_F  64
#define OUT_F  40
#define ROWT   4      // 16-row tiles per block

typedef __bf16 bf16_t;
typedef __attribute__((ext_vector_type(16))) __bf16 v16bf;
typedef __attribute__((ext_vector_type(8)))  __bf16 v8bf;
typedef __attribute__((ext_vector_type(4)))  __bf16 v4bf;
typedef __attribute__((ext_vector_type(8)))  float  v8f;

__device__ __forceinline__ v16bf cat8(v8bf lo, v8bf hi) {
    return __builtin_shufflevector(lo, hi, 0,1,2,3,4,5,6,7,8,9,10,11,12,13,14,15);
}
__device__ __forceinline__ v4bf cvt4(float4 v) {
    v4bf r; r[0] = (bf16_t)v.x; r[1] = (bf16_t)v.y; r[2] = (bf16_t)v.z; r[3] = (bf16_t)v.w;
    return r;
}

// ---------------------------------------------------------------- degree
__global__ void gcn_degree(const int* __restrict__ dst, float* __restrict__ deg, int nE) {
    int e = blockIdx.x * blockDim.x + threadIdx.x;
    if (e < nE) atomicAdd(&deg[dst[e]], 1.0f);
}

// deg -> dinv = rsqrt(deg + 1)   (self-loop contributes the +1; always > 0)
__global__ void gcn_rsqrt(float* __restrict__ d, int n) {
    int i = blockIdx.x * blockDim.x + threadIdx.x;
    if (i < n) d[i] = rsqrtf(d[i] + 1.0f);
}

// ---------------------------------------------------------------- GEMM1: h1 = x @ W1  (N x 128) * (128 x 64)
// 64 rows per block (4 tiles of 16); 4 waves, each owns one 16-col tile.
__global__ __launch_bounds__(128) void gcn_gemm1_wmma(
    const float* __restrict__ x, const float* __restrict__ W1,
    float* __restrict__ h1, int n) {
    __shared__ alignas(64) bf16_t sWB[4][4][32][16];     // B fragments, 16 KB
    __shared__ alignas(64) bf16_t sX[16][IN_F + 8];      // A tile
    __shared__ alignas(16) float  sC[16][HID_F + 4];     // C tile for coalesced store

    const int tid  = threadIdx.x;
    const int wave = tid >> 5;
    const int lane = tid & 31;

    // stage W1 pre-swizzled: fragment(lane,t) = W1[kk*32 + 8*(lane>>4) + (t&7) + 16*(t>>3)][ct*16 + (lane&15)]
    for (int i = tid; i < 4 * 4 * 32 * 16; i += 128) {
        int t  = i & 15;
        int ln = (i >> 4) & 31;
        int ct = (i >> 9) & 3;
        int kk = i >> 11;
        int k  = kk * 32 + 8 * (ln >> 4) + (t & 7) + 16 * (t >> 3);
        int c  = ct * 16 + (ln & 15);
        sWB[kk][ct][ln][t] = (bf16_t)W1[k * HID_F + c];
    }
    __syncthreads();

    // hoist this wave's 4 B fragments into registers (2x ds_load_b128 each)
    const v16bf b0 = *(const v16bf*)sWB[0][wave][lane];
    const v16bf b1 = *(const v16bf*)sWB[1][wave][lane];
    const v16bf b2 = *(const v16bf*)sWB[2][wave][lane];
    const v16bf b3 = *(const v16bf*)sWB[3][wave][lane];

    const int m       = lane & 15;
    const int khalf   = lane >> 4;        // 0: lanes 0-15, 1: lanes 16-31
    const int col     = wave * 16 + m;
    const int rowbase = blockIdx.x * (16 * ROWT);
    const bool full   = (rowbase + 16 * ROWT) <= n;   // block-uniform fast path

    for (int rt = 0; rt < ROWT; ++rt) {
        const int row0 = rowbase + rt * 16;
        if (row0 >= n) break;             // uniform per block
        if (full) {
            // vectorized fill: 512 float4 chunks, 4 per thread
            for (int i = tid; i < 512; i += 128) {
                int r = i >> 5, c4 = (i & 31) << 2;
                float4 v = *(const float4*)&x[(row0 + r) * IN_F + c4];
                *(v4bf*)&sX[r][c4] = cvt4(v);
            }
        } else {
            for (int i = tid; i < 16 * IN_F; i += 128) {
                int r = i >> 7, cc = i & 127;
                int gr = row0 + r;
                sX[r][cc] = (gr < n) ? (bf16_t)x[gr * IN_F + cc] : (bf16_t)0.f;
            }
        }
        __syncthreads();

        v8f acc = {};
#pragma unroll
        for (int kk = 0; kk < 4; ++kk) {
            const int kb = kk * 32;
            // A (16x32 bf16): halves 0-7 -> K = kb+8*khalf+t ; 8-15 -> K = kb+16+8*khalf+t
            v8bf alo = *(const v8bf*)&sX[m][kb + 8 * khalf];
            v8bf ahi = *(const v8bf*)&sX[m][kb + 16 + 8 * khalf];
            v16bf a  = cat8(alo, ahi);
            v16bf b  = (kk == 0) ? b0 : (kk == 1) ? b1 : (kk == 2) ? b2 : b3;
            acc = __builtin_amdgcn_wmma_f32_16x16x32_bf16(false, a, false, b,
                                                          (short)0, acc, false, false);
        }
        // C/D layout: acc[r] = out[r + 8*khalf][col] -> stage in LDS
#pragma unroll
        for (int r = 0; r < 8; ++r) sC[8 * khalf + r][col] = acc[r];
        __syncthreads();

        // coalesced store: 16 rows x 16 float4 chunks = 256, 2 per thread
        for (int i = tid; i < 16 * (HID_F / 4); i += 128) {
            int r = i >> 4, c4 = (i & 15) << 2;
            int gr = row0 + r;
            if (full || gr < n)
                *(float4*)&h1[gr * HID_F + c4] = *(const float4*)&sC[r][c4];
        }
        __syncthreads();
    }
}

// ---------------------------------------------------------------- GEMM2: h2 = h1r @ W2  (N x 64) * (64 x 40), cols padded to 48
__global__ __launch_bounds__(96) void gcn_gemm2_wmma(
    const float* __restrict__ hin, const float* __restrict__ W2,
    float* __restrict__ h2, int n) {
    __shared__ alignas(64) bf16_t sWB[2][3][32][16];     // 6 KB
    __shared__ alignas(64) bf16_t sX[16][HID_F + 8];
    __shared__ alignas(16) float  sC[16][48 + 4];

    const int tid  = threadIdx.x;       // 0..95 -> 3 waves
    const int wave = tid >> 5;
    const int lane = tid & 31;

    for (int i = tid; i < 2 * 3 * 32 * 16; i += 96) {
        int t    = i & 15;
        int ln   = (i >> 4) & 31;
        int rest = i >> 9;              // 0..5
        int ct   = rest % 3;
        int kk   = rest / 3;
        int k    = kk * 32 + 8 * (ln >> 4) + (t & 7) + 16 * (t >> 3);
        int c    = ct * 16 + (ln & 15);
        sWB[kk][ct][ln][t] = (c < OUT_F) ? (bf16_t)W2[k * OUT_F + c] : (bf16_t)0.f;
    }
    __syncthreads();

    const v16bf b0 = *(const v16bf*)sWB[0][wave][lane];
    const v16bf b1 = *(const v16bf*)sWB[1][wave][lane];

    const int m       = lane & 15;
    const int khalf   = lane >> 4;
    const int col     = wave * 16 + m;
    const int rowbase = blockIdx.x * (16 * ROWT);
    const bool full   = (rowbase + 16 * ROWT) <= n;

    for (int rt = 0; rt < ROWT; ++rt) {
        const int row0 = rowbase + rt * 16;
        if (row0 >= n) break;
        if (full) {
            for (int i = tid; i < 256; i += 96) {   // 16 rows x 16 float4
                int r = i >> 4, c4 = (i & 15) << 2;
                float4 v = *(const float4*)&hin[(row0 + r) * HID_F + c4];
                *(v4bf*)&sX[r][c4] = cvt4(v);
            }
        } else {
            for (int i = tid; i < 16 * HID_F; i += 96) {
                int r = i >> 6, cc = i & 63;
                int gr = row0 + r;
                sX[r][cc] = (gr < n) ? (bf16_t)hin[gr * HID_F + cc] : (bf16_t)0.f;
            }
        }
        __syncthreads();

        v8f acc = {};
#pragma unroll
        for (int kk = 0; kk < 2; ++kk) {
            const int kb = kk * 32;
            v8bf alo = *(const v8bf*)&sX[m][kb + 8 * khalf];
            v8bf ahi = *(const v8bf*)&sX[m][kb + 16 + 8 * khalf];
            v16bf a  = cat8(alo, ahi);
            v16bf b  = (kk == 0) ? b0 : b1;
            acc = __builtin_amdgcn_wmma_f32_16x16x32_bf16(false, a, false, b,
                                                          (short)0, acc, false, false);
        }
#pragma unroll
        for (int r = 0; r < 8; ++r) sC[8 * khalf + r][col] = acc[r];
        __syncthreads();

        // store only the 40 valid columns: 16 rows x 10 float4 chunks
        for (int i = tid; i < 16 * (OUT_F / 4); i += 96) {
            int r = i / 10, c4 = (i % 10) << 2;
            int gr = row0 + r;
            if (full || gr < n)
                *(float4*)&h2[gr * OUT_F + c4] = *(const float4*)&sC[r][c4];
        }
        __syncthreads();
    }
}

// ---------------------------------------------------------------- edge gather+scale+scatter (1 thread = 1 edge x 4 feats)
__global__ void gcn_aggregate(const float* __restrict__ feat,
                              const int*   __restrict__ src,
                              const int*   __restrict__ dst,
                              const float* __restrict__ dinv,
                              float*       __restrict__ agg,
                              int nE, int F, int chunks) {
    int gid = blockIdx.x * blockDim.x + threadIdx.x;
    int total = nE * chunks;
    if (gid >= total) return;
    int e = gid / chunks;
    int f = (gid % chunks) * 4;
    int s = src[e], d = dst[e];
    float nrm = dinv[s] * dinv[d];
    const float4 v = *(const float4*)(feat + s * F + f);
    float* ap = agg + d * F + f;
    atomicAdd(ap + 0, v.x * nrm);
    atomicAdd(ap + 1, v.y * nrm);
    atomicAdd(ap + 2, v.z * nrm);
    atomicAdd(ap + 3, v.w * nrm);
}

// add self-loop term + bias, ReLU; in-place on agg
__global__ void gcn_finish1(const float* __restrict__ h1,
                            const float* __restrict__ dinv,
                            const float* __restrict__ b1,
                            float* agg, int n) {
    int gid = blockIdx.x * blockDim.x + threadIdx.x;
    if (gid >= n * HID_F) return;
    int i = gid >> 6, f = gid & 63;
    float di = dinv[i];
    float v = agg[gid] + h1[gid] * di * di + b1[f];
    agg[gid] = v > 0.f ? v : 0.f;
}

// add self-loop term + bias to the already-scattered output
__global__ void gcn_finish2(const float* __restrict__ h2,
                            const float* __restrict__ dinv,
                            const float* __restrict__ b2,
                            float* out, int n) {
    int gid = blockIdx.x * blockDim.x + threadIdx.x;
    if (gid >= n * OUT_F) return;
    int i = gid / OUT_F, f = gid % OUT_F;
    float di = dinv[i];
    out[gid] += h2[gid] * di * di + b2[f];
}

// ----------------------------------------------------------------
extern "C" void kernel_launch(void* const* d_in, const int* in_sizes, int n_in,
                              void* d_out, int out_size, void* d_ws, size_t ws_size,
                              hipStream_t stream) {
    const float* x   = (const float*)d_in[0];
    const int*   ei  = (const int*)d_in[1];     // [2, E] row-major: src then dst
    const float* W1  = (const float*)d_in[2];
    const float* b1  = (const float*)d_in[3];
    const float* W2  = (const float*)d_in[4];
    const float* b2  = (const float*)d_in[5];
    float* out = (float*)d_out;

    const int n  = in_sizes[0] / IN_F;
    const int nE = in_sizes[1] / 2;
    const int* srcI = ei;
    const int* dstI = ei + nE;

    // workspace carve-up (256B aligned)
    char* ws = (char*)d_ws;
    size_t off = 0;
    auto carve = [&](size_t bytes) {
        void* p = ws + off;
        off = (off + bytes + 255) & ~(size_t)255;
        return p;
    };
    float* dinv = (float*)carve((size_t)n * sizeof(float));
    float* h1   = (float*)carve((size_t)n * HID_F * sizeof(float));
    float* agg1 = (float*)carve((size_t)n * HID_F * sizeof(float));
    float* h2   = (float*)carve((size_t)n * OUT_F * sizeof(float));
    (void)ws_size;

    hipMemsetAsync(dinv, 0, (size_t)n * sizeof(float), stream);
    hipMemsetAsync(agg1, 0, (size_t)n * HID_F * sizeof(float), stream);
    hipMemsetAsync(out,  0, (size_t)n * OUT_F * sizeof(float), stream);

    // normalization
    gcn_degree<<<(nE + 255) / 256, 256, 0, stream>>>(dstI, dinv, nE);
    gcn_rsqrt <<<(n + 255) / 256, 256, 0, stream>>>(dinv, n);

    const int rowsPerBlock = 16 * ROWT;

    // layer 1
    gcn_gemm1_wmma<<<(n + rowsPerBlock - 1) / rowsPerBlock, 128, 0, stream>>>(x, W1, h1, n);
    {
        int total = nE * (HID_F / 4);
        gcn_aggregate<<<(total + 255) / 256, 256, 0, stream>>>(
            h1, srcI, dstI, dinv, agg1, nE, HID_F, HID_F / 4);
    }
    {
        int total = n * HID_F;
        gcn_finish1<<<(total + 255) / 256, 256, 0, stream>>>(h1, dinv, b1, agg1, n);
    }

    // layer 2
    gcn_gemm2_wmma<<<(n + rowsPerBlock - 1) / rowsPerBlock, 96, 0, stream>>>(agg1, W2, h2, n);
    {
        int total = nE * (OUT_F / 4);
        gcn_aggregate<<<(total + 255) / 256, 256, 0, stream>>>(
            h2, srcI, dstI, dinv, out, nE, OUT_F, OUT_F / 4);
    }
    {
        int total = n * OUT_F;
        gcn_finish2<<<(total + 255) / 256, 256, 0, stream>>>(h2, dinv, b2, out, n);
    }
}
